// CrossDeformableAttention_35201551958440
// MI455X (gfx1250) — matrix operand order repaired and checked by reading, not compiled, not run on hardware
//
#include <hip/hip_runtime.h>
#include <hip/hip_bf16.h>

#define DIMC 256
#define NHh  8
#define NPp  8
#define HDd  32
#define NQ   4096
#define NB   4
#define MTOT (NB * NQ)   // 16384

typedef __attribute__((ext_vector_type(16))) __bf16 v16bf;
typedef __attribute__((ext_vector_type(8)))  __bf16 v8bf;
typedef __attribute__((ext_vector_type(8)))  float  v8f;

#if defined(__gfx1250__) && \
    __has_builtin(__builtin_amdgcn_global_load_async_to_lds_b32) && \
    __has_builtin(__builtin_amdgcn_s_wait_asynccnt)
#define USE_ASYNC_LDS 1
typedef __attribute__((address_space(1))) int* gptr_i32;
typedef __attribute__((address_space(3))) int* lptr_i32;
#else
#define USE_ASYNC_LDS 0
#endif

// ---------------------------------------------------------------------------
// Kernel 0: pack weights to bf16 (row-major, rows = output dim n). This is the
// exact B-fragment layout feed: lane n = l&15 reads 16 contiguous K values.
// ---------------------------------------------------------------------------
__global__ void pack_kernel(const float* __restrict__ Wloc, const float* __restrict__ bloc,
                            const float* __restrict__ Watt, const float* __restrict__ batt,
                            const float* __restrict__ Wval, const float* __restrict__ Wout,
                            __bf16* __restrict__ Wqa, float* __restrict__ bqa,
                            __bf16* __restrict__ Wvp, __bf16* __restrict__ Wop) {
  int tid = blockIdx.x * blockDim.x + threadIdx.x;
  int stride = gridDim.x * blockDim.x;
  for (int i = tid; i < 192 * 256; i += stride) {
    int n = i >> 8, k = i & 255;
    float v = (n < 128) ? Wloc[(size_t)n * 256 + k] : Watt[(size_t)(n - 128) * 256 + k];
    Wqa[i] = (__bf16)v;
  }
  for (int i = tid; i < 256 * 256; i += stride) Wvp[i] = (__bf16)Wval[i];
  for (int i = tid; i < 256 * 256; i += stride) Wop[i] = (__bf16)Wout[i];
  for (int i = tid; i < 192; i += stride) bqa[i] = (i < 128) ? bloc[i] : batt[i - 128];
}

// ---------------------------------------------------------------------------
// Kernels 1/2: projection GEMM.  A is [B, DIM, NQ] in memory (i.e. A^T of the
// logical [M=16384, K=256] matrix).  One workgroup = 4 waves, M-tile 16.
// The A tile is staged ONCE in LDS (async global->LDS copy when available),
// then each wave loops over its share of all N-tiles, so query/value are read
// from HBM exactly once.  A and B fragments both use contiguous k = 16*(lane/16) + j.
//   is_val==0: outputs n<128 -> loc buffer, n in [128,192) -> att logits
//   is_val==1: outputs scattered into vimg[b][h][y][x][d] (d contiguous)
// ---------------------------------------------------------------------------
__global__ void proj_gemm_kernel(const float* __restrict__ A,     // [B, DIM, NQ]
                                 const __bf16* __restrict__ Wp,   // [Ntot, 256] bf16
                                 const float* __restrict__ bias,  // [Ntot]
                                 float* __restrict__ locbuf,      // [MTOT,128]
                                 float* __restrict__ attbuf,      // [MTOT,64]
                                 float* __restrict__ vimg,        // [B,NH,NQ,HD]
                                 int ntiles, int is_val) {
  __shared__ float Alds[16 * 260];
  const int t  = threadIdx.x;
  const int m0 = blockIdx.x * 16;
  const int bb = m0 >> 12;
  const int mq = m0 & (NQ - 1);
  const float* Ab = A + (size_t)bb * DIMC * NQ + mq;

#if USE_ASYNC_LDS
#pragma unroll
  for (int r = 0; r < 32; ++r) {          // 128 threads x 32 = 4096 elements
    int e  = r * 128 + t;
    int k  = e >> 4;
    int ml = e & 15;                       // 16 consecutive m contiguous in mem
    const float* gp = Ab + (size_t)k * NQ + ml;
    __builtin_amdgcn_global_load_async_to_lds_b32(
        (gptr_i32)(unsigned long long)(uintptr_t)gp,
        (lptr_i32)(unsigned)(uintptr_t)(&Alds[ml * 260 + k]),
        0, 0);
  }
  __builtin_amdgcn_s_wait_asynccnt(0);
#else
#pragma unroll
  for (int r = 0; r < 32; ++r) {
    int e  = r * 128 + t;
    int k  = e >> 4;
    int ml = e & 15;
    Alds[ml * 260 + k] = Ab[(size_t)k * NQ + ml];
  }
#endif
  __syncthreads();

  const int wave    = t >> 5;
  const int lane    = t & 31;
  const int nl      = lane & 15;
  const int halfsel = lane >> 4;
  // A fragment: lane row m = nl, K = k0 + halfsel*16 + j   (j = 0..15)
  const float* arow = Alds + nl * 260 + halfsel * 16;

  for (int nt = wave; nt < ntiles; nt += 4) {
    const int n0 = nt * 16;
    const int n  = n0 + nl;

    v8f c;
#pragma unroll
    for (int i = 0; i < 8; ++i) c[i] = 0.f;

    // B fragment: lane col n, K = k0 + halfsel*16 + j   (j = 0..15)
    const __bf16* wrow = Wp + (size_t)n * 256 + halfsel * 16;

#pragma unroll
    for (int k0 = 0; k0 < 256; k0 += 32) {
      __builtin_prefetch(wrow + k0 + 64, 0, 0);   // global_prefetch_b8
      v16bf bf = *(const v16bf*)(wrow + k0);
      v16bf af;
#pragma unroll
      for (int j = 0; j < 8; ++j) {
        af[j]     = (__bf16)arow[k0 + j];
        af[8 + j] = (__bf16)arow[k0 + 8 + j];
      }
      c = __builtin_amdgcn_wmma_f32_16x16x32_bf16(false, af, false, bf,
                                                  (short)0, c, false, false);
    }

    // D fragment: lane col n = nl, rows M = halfsel*8 + i
    const float bn = bias[n];
    const int   mb = halfsel * 8;
    if (!is_val) {
#pragma unroll
      for (int i = 0; i < 8; ++i) {
        int m = m0 + mb + i;
        float v = c[i] + bn;
        if (n < 128) locbuf[(size_t)m * 128 + n] = v;
        else         attbuf[(size_t)m * 64 + (n - 128)] = v;
      }
    } else {
      int hh = n >> 5, dd = n & 31;
#pragma unroll
      for (int i = 0; i < 8; ++i) {
        int m = m0 + mb + i;
        int b2 = m >> 12, q2 = m & (NQ - 1);
        vimg[((size_t)(b2 * NHh + hh) * NQ + q2) * HDd + dd] = c[i] + bn;
      }
    }
  }
}

// ---------------------------------------------------------------------------
// Kernel 3: fused softmax + bilinear sampling.  One wave per (query, head),
// lane = channel d.  vimg has channels contiguous -> each corner tap is a
// coalesced 128B read across the wave.  Writes bf16 row-major [MTOT, 256],
// which is the exact A-fragment feed for the output GEMM.
// ---------------------------------------------------------------------------
__global__ void msda_sample_kernel(const float* __restrict__ locbuf,
                                   const float* __restrict__ attbuf,
                                   const float* __restrict__ vimg,
                                   __bf16* __restrict__ msda) {
  const int lane = threadIdx.x & 31;                 // channel d
  const int wid  = blockIdx.x * 8 + (threadIdx.x >> 5);
  const int m = wid >> 3;
  const int h = wid & 7;
  const int b = m >> 12;

  const float* lp = locbuf + (size_t)m * 128 + h * 16;  // 8 points x (x,y)
  const float* ap = attbuf + (size_t)m * 64 + h * 8;    // 8 logits

  float lg[8];
  float mx = -1e30f;
#pragma unroll
  for (int p = 0; p < 8; ++p) { lg[p] = ap[p]; mx = fmaxf(mx, lg[p]); }
  float s = 0.f;
#pragma unroll
  for (int p = 0; p < 8; ++p) { lg[p] = __expf(lg[p] - mx); s += lg[p]; }
  const float inv = 1.f / s;

  const float* vb = vimg + (size_t)(b * NHh + h) * NQ * HDd + lane;
  float acc = 0.f;
#pragma unroll
  for (int p = 0; p < 8; ++p) {
    // grid_sample: x = (2*loc-1 + 1)*W/2 - 0.5 = loc*64 - 0.5
    float x = lp[p * 2 + 0] * 64.f - 0.5f;
    float y = lp[p * 2 + 1] * 64.f - 0.5f;
    float x0f = floorf(x), y0f = floorf(y);
    float wx = x - x0f, wy = y - y0f;
    int x0 = (int)x0f, y0 = (int)y0f;
    bool xi0 = (x0 >= 0) && (x0 < 64), xi1 = (x0 + 1 >= 0) && (x0 + 1 < 64);
    bool yi0 = (y0 >= 0) && (y0 < 64), yi1 = (y0 + 1 >= 0) && (y0 + 1 < 64);
    float v00 = 0.f, v01 = 0.f, v10 = 0.f, v11 = 0.f;
    if (xi0 && yi0) v00 = vb[((size_t)(y0 * 64 + x0)) * HDd];
    if (xi1 && yi0) v01 = vb[((size_t)(y0 * 64 + x0 + 1)) * HDd];
    if (xi0 && yi1) v10 = vb[((size_t)((y0 + 1) * 64 + x0)) * HDd];
    if (xi1 && yi1) v11 = vb[((size_t)((y0 + 1) * 64 + x0 + 1)) * HDd];
    float top = v00 + wx * (v01 - v00);
    float bot = v10 + wx * (v11 - v10);
    acc += (lg[p] * inv) * (top + wy * (bot - top));
  }
  msda[(size_t)m * 256 + h * HDd + lane] = (__bf16)acc;
}

// ---------------------------------------------------------------------------
// Kernel 4: output GEMM.  A (msda, bf16 row-major) fragments loaded straight
// from global: two 16B half-fragment loads per k-step.  Bias epilogue to f32.
// ---------------------------------------------------------------------------
__global__ void out_gemm_kernel(const __bf16* __restrict__ Abf,  // [MTOT,256]
                                const __bf16* __restrict__ Wp,   // [256,256]
                                const float* __restrict__ bias,  // [256]
                                float* __restrict__ out) {
  const int t = threadIdx.x;
  const int wave = t >> 5, lane = t & 31;
  const int nl = lane & 15, halfsel = lane >> 4;
  const int m0 = blockIdx.x * 16;
  const int n0 = (blockIdx.y * 4 + wave) * 16;
  const int n  = n0 + nl;

  v8f c;
#pragma unroll
  for (int i = 0; i < 8; ++i) c[i] = 0.f;

  const __bf16* arow = Abf + (size_t)(m0 + nl) * 256 + halfsel * 16;
  const __bf16* wrow = Wp + (size_t)n * 256 + halfsel * 16;
#pragma unroll
  for (int k0 = 0; k0 < 256; k0 += 32) {
    __builtin_prefetch(arow + k0 + 64, 0, 0);
    v8bf lo = *(const v8bf*)(arow + k0);
    v8bf hi = *(const v8bf*)(arow + k0 + 8);
    v16bf af;
#pragma unroll
    for (int j = 0; j < 8; ++j) { af[j] = lo[j]; af[8 + j] = hi[j]; }
    v16bf bf = *(const v16bf*)(wrow + k0);
    c = __builtin_amdgcn_wmma_f32_16x16x32_bf16(false, af, false, bf,
                                                (short)0, c, false, false);
  }
  const float bn = bias[n];
  const int   mb = halfsel * 8;
#pragma unroll
  for (int i = 0; i < 8; ++i)
    out[(size_t)(m0 + mb + i) * 256 + n] = c[i] + bn;
}

// ---------------------------------------------------------------------------
extern "C" void kernel_launch(void* const* d_in, const int* in_sizes, int n_in,
                              void* d_out, int out_size, void* d_ws, size_t ws_size,
                              hipStream_t stream) {
  (void)in_sizes; (void)n_in; (void)out_size; (void)ws_size;
  const float* query = (const float*)d_in[0];  // [B, DIM, H, W]
  const float* value = (const float*)d_in[1];
  const float* Wloc  = (const float*)d_in[2];
  const float* bloc  = (const float*)d_in[3];
  const float* Watt  = (const float*)d_in[4];
  const float* batt  = (const float*)d_in[5];
  const float* Wval  = (const float*)d_in[6];
  const float* bval  = (const float*)d_in[7];
  const float* Wout  = (const float*)d_in[8];
  const float* bout  = (const float*)d_in[9];
  float* out = (float*)d_out;

  char* ws = (char*)d_ws;
  size_t off = 0;
  auto take = [&](size_t bytes) -> char* {
    char* p = ws + off;
    off = (off + bytes + 255) & ~(size_t)255;
    return p;
  };
  __bf16* Wqa    = (__bf16*)take((size_t)192 * 256 * 2);
  __bf16* Wvp    = (__bf16*)take((size_t)256 * 256 * 2);
  __bf16* Wop    = (__bf16*)take((size_t)256 * 256 * 2);
  float*  bqa    = (float*)take((size_t)192 * 4);
  float*  locbuf = (float*)take((size_t)MTOT * 128 * 4);
  float*  attbuf = (float*)take((size_t)MTOT * 64 * 4);
  float*  vimg   = (float*)take((size_t)NB * NHh * NQ * HDd * 4);
  __bf16* msda   = (__bf16*)take((size_t)MTOT * 256 * 2);

  pack_kernel<<<128, 256, 0, stream>>>(Wloc, bloc, Watt, batt, Wval, Wout,
                                       Wqa, bqa, Wvp, Wop);
  // loc + att logits: N = 192 -> 12 N-tiles, A staged once per M-tile
  proj_gemm_kernel<<<MTOT / 16, 128, 0, stream>>>(
      query, Wqa, bqa, locbuf, attbuf, nullptr, 12, 0);
  // value projection: N = 256 -> 16 N-tiles, scattered into vimg layout
  proj_gemm_kernel<<<MTOT / 16, 128, 0, stream>>>(
      value, Wvp, bval, nullptr, nullptr, vimg, 16, 1);
  // sampling: 16384*8 (query,head) waves, 8 waves per block
  msda_sample_kernel<<<MTOT * NHh / 8, 256, 0, stream>>>(locbuf, attbuf, vimg, msda);
  // output projection
  out_gemm_kernel<<<dim3(MTOT / 16, 4), 128, 0, stream>>>(msda, Wop, bout, out);
}
